// ResonantRouter_63934883168714
// MI455X (gfx1250) — matrix operand order repaired and checked by reading, not compiled, not run on hardware
//
#include <hip/hip_runtime.h>
#include <hip/hip_bf16.h>
#include <stdint.h>

// ---------------------------------------------------------------------------
// ResonantRouter on gfx1250: complex phasor iteration as bf16 WMMA GEMMs.
// B=4,T=2048 -> BT=8192 tokens; H=8 heads; K=512 buckets; D=256.
// fp32 master state in registers, bf16 mirror in LDS (A-frag = 2x ds_load_b128).
// Pattern fragments precomputed in fragment-contiguous bf16 buffers; all
// fragment addresses wave-uniform (readfirstlane) -> scalar addressing.
// WMMA issued in independent-accumulator passes to kill hazard NOPs.
// ---------------------------------------------------------------------------

typedef __attribute__((ext_vector_type(16))) __bf16 v16bf;
typedef __attribute__((ext_vector_type(8)))  float  v8f;

#define BT      8192
#define Hh      8
#define Kk      512
#define Dd      256
#define STEPS   20
#define TM      16          // tokens per block
#define FRAGB   1024        // bytes per 16x32/32x16 bf16 fragment (32 lanes * 32B)
#define PARTB   ((size_t)2048 * FRAGB)  // 2 MiB per pattern buffer part

union Frag {
    uint32_t u[8];
    uint4    q[2];
    v16bf    b;
};

// cheap fp32 -> bf16 (round-half-up)
static __device__ inline uint32_t f2bf(float f) {
    return (__float_as_uint(f) + 0x8000u) >> 16;
}
static __device__ inline uint32_t pack2bf(float lo, float hi) {
    return (f2bf(lo) & 0xFFFFu) | (f2bf(hi) << 16);
}
static __device__ inline v8f wmma_bf16(v16bf a, v16bf b, v8f c) {
    return __builtin_amdgcn_wmma_f32_16x16x32_bf16(
        false, a, false, b, (short)0, c, false, false);
}
static __device__ inline v8f zero8() {
    v8f z;
#pragma unroll
    for (int i = 0; i < 8; ++i) z[i] = 0.0f;
    return z;
}
static __device__ inline float clean(float v) {
    v = isnan(v) ? 0.0f : v;
    return fminf(fmaxf(v, -100.0f), 100.0f);
}
static __device__ inline void load_frag(Frag& F, const uint8_t* p, int laneB) {
    F.q[0] = *(const uint4*)(p + laneB);
    F.q[1] = *(const uint4*)(p + laneB + 16);
}

// ---------------------------------------------------------------------------
// Prep: patterns (H,K,D) fp32 -> clip -> cos/sin -> bf16 fragment buffers.
// buf 0: PT cos (GEMM1 B-operand: rows=d in 32-chunk, cols=k in 16-tile)
// buf 1: PT sin
// buf 2: PN cos (GEMM2 B-operand: rows=k in 32-chunk, cols=d in 16-tile)
// buf 3: PN sin
// ---------------------------------------------------------------------------
__global__ __launch_bounds__(256) void rr_prep(const float* __restrict__ pat,
                                               uint8_t* __restrict__ ws) {
    int gid  = blockIdx.x * 256 + threadIdx.x;
    int lane = gid & 31;
    int f    = gid >> 5;
    int buf  = f >> 11;          // 0..3
    int fi   = f & 2047;
    int half = lane >> 4;
    int l16  = lane & 15;

    uint32_t* dst = (uint32_t*)(ws + (size_t)buf * PARTB + (size_t)fi * FRAGB + lane * 32);
#pragma unroll 1
    for (int v = 0; v < 8; ++v) {
        float vals[2];
#pragma unroll
        for (int e = 0; e < 2; ++e) {
            int row = 2 * v + e + 8 * half + (v >= 4 ? 8 : 0);   // B-operand row (WMMA k-dim)
            int kIdx, dIdx, hh;
            if (buf < 2) {                     // PT: fi = (h*8+dc)*32 + kt
                int kt = fi & 31, dc = (fi >> 5) & 7; hh = fi >> 8;
                kIdx = kt * 16 + l16;
                dIdx = dc * 32 + row;
            } else {                           // PN: fi = (h*16+kc)*16 + dt
                int dt = fi & 15, kc = (fi >> 4) & 15; hh = fi >> 8;
                kIdx = kc * 32 + row;
                dIdx = dt * 16 + l16;
            }
            float p = pat[((size_t)hh * Kk + kIdx) * Dd + dIdx];
            p = fminf(fmaxf(p, -100.0f), 100.0f);
            vals[e] = ((buf & 1) == 0) ? __cosf(p) : __sinf(p);
        }
        dst[v] = pack2bf(vals[0], vals[1]);
    }
}

// ---------------------------------------------------------------------------
// Main: one block = 16 tokens x one head; state resident for all 20 steps.
// ---------------------------------------------------------------------------
__global__ __launch_bounds__(256) void rr_main(const float* __restrict__ tag,
                                               const uint8_t* __restrict__ ptA,
                                               const uint8_t* __restrict__ ptB,
                                               const uint8_t* __restrict__ pnA,
                                               const uint8_t* __restrict__ pnB,
                                               float* __restrict__ out) {
    __shared__ unsigned short Xbf[TM * Dd];   // 8 KB  bf16 mirror of x
    __shared__ unsigned short Ybf[TM * Dd];   // 8 KB  bf16 mirror of y
    __shared__ unsigned short Urs[TM * Kk];   // 16 KB bf16 u_r
    __shared__ unsigned short Uis[TM * Kk];   // 16 KB bf16 u_i

    const int lane  = threadIdx.x & 31;
    const int wave  = __builtin_amdgcn_readfirstlane(threadIdx.x >> 5); // uniform
    const int half  = lane >> 4;
    const int l16   = lane & 15;
    const int laneB = lane * 32;              // per-lane fragment byte offset
    const int h     = blockIdx.y;
    const int tokBase = blockIdx.x * TM;

    const float invD  = 1.0f / (float)Dd;
    const float scale = 0.25f / (float)Kk;    // COUPLING / K
    const float damp  = 1.0f - 0.02f;         // 1 - DAMPING

    // ---- fp32 master state in registers, at update-phase (C-frag) slots ----
    // slot (j,v): row m = v + 8*half, col d = (wave*2+j)*16 + l16
    float xr[16], yr[16];
#pragma unroll
    for (int j = 0; j < 2; ++j) {
        int d = (wave * 2 + j) * 16 + l16;
#pragma unroll
        for (int v = 0; v < 8; ++v) {
            int m = v + 8 * half;
            float t = clean(tag[(size_t)(tokBase + m) * Dd + d]);
            xr[j * 8 + v] = __cosf(t);
            yr[j * 8 + v] = __sinf(t);
        }
    }
    // ---- bf16 mirror in LDS (cooperative, row-major [TM][Dd]) ----
#pragma unroll
    for (int i = 0; i < (TM * Dd) / 256; ++i) {
        int idx = threadIdx.x + i * 256;
        int m = idx >> 8, d = idx & 255;
        float t = clean(tag[(size_t)(tokBase + m) * Dd + d]);
        Xbf[idx] = (unsigned short)f2bf(__cosf(t));
        Ybf[idx] = (unsigned short)f2bf(__sinf(t));
    }
    __syncthreads();

#pragma unroll 1
    for (int step = 0; step < STEPS; ++step) {
        // ================= GEMM1: U = (x+iy) @ conj-proj  =================
        v8f ur[4], ui[4];
#pragma unroll
        for (int j = 0; j < 4; ++j) { ur[j] = zero8(); ui[j] = zero8(); }

#pragma unroll 1
        for (int dc = 0; dc < 8; ++dc) {
            int base = l16 * Dd + dc * 32 + half * 8;    // A-frag: 2x b128
            Frag FX, FY, FYn;
            FX.q[0] = *(const uint4*)&Xbf[base];
            FX.q[1] = *(const uint4*)&Xbf[base + 16];
            FY.q[0] = *(const uint4*)&Ybf[base];
            FY.q[1] = *(const uint4*)&Ybf[base + 16];
#pragma unroll
            for (int v = 0; v < 8; ++v) FYn.u[v] = FY.u[v] ^ 0x80008000u;

            Frag PA[4], PB[4];
#pragma unroll
            for (int j = 0; j < 4; ++j) {
                // wave-uniform fragment offset -> scalar addressing
                size_t uoff = ((size_t)((h * 8 + dc) * 32 + wave * 4 + j)) * FRAGB;
                load_frag(PA[j], ptA + uoff, laneB);
                load_frag(PB[j], ptB + uoff, laneB);
            }
            // independent-accumulator passes (>=4 WMMA between dependents)
#pragma unroll
            for (int j = 0; j < 4; ++j) ur[j] = wmma_bf16(FX.b,  PA[j].b, ur[j]);
#pragma unroll
            for (int j = 0; j < 4; ++j) ui[j] = wmma_bf16(FY.b,  PA[j].b, ui[j]);
#pragma unroll
            for (int j = 0; j < 4; ++j) ur[j] = wmma_bf16(FYn.b, PB[j].b, ur[j]);
#pragma unroll
            for (int j = 0; j < 4; ++j) ui[j] = wmma_bf16(FX.b,  PB[j].b, ui[j]);
        }
        // store U to LDS (row-major bf16) -> GEMM2 A-frags are 2x ds_load_b128
#pragma unroll
        for (int j = 0; j < 4; ++j) {
            int col = (wave * 4 + j) * 16 + l16;
#pragma unroll
            for (int v = 0; v < 8; ++v) {
                int m = v + 8 * half;
                Urs[m * Kk + col] = (unsigned short)f2bf(ur[j][v]);
                Uis[m * Kk + col] = (unsigned short)f2bf(ui[j][v]);
            }
        }
        __syncthreads();

        // ================= GEMM2: V = U @ proj  ===========================
        v8f vr[2], vi[2];
#pragma unroll
        for (int j = 0; j < 2; ++j) { vr[j] = zero8(); vi[j] = zero8(); }

#pragma unroll 1
        for (int kc = 0; kc < 16; ++kc) {
            int base = l16 * Kk + kc * 32 + half * 8;
            Frag FUr, FUi, FUrn;
            FUr.q[0] = *(const uint4*)&Urs[base];
            FUr.q[1] = *(const uint4*)&Urs[base + 16];
            FUi.q[0] = *(const uint4*)&Uis[base];
            FUi.q[1] = *(const uint4*)&Uis[base + 16];
#pragma unroll
            for (int v = 0; v < 8; ++v) FUrn.u[v] = FUr.u[v] ^ 0x80008000u;

            Frag PA[2], PB[2];
#pragma unroll
            for (int j = 0; j < 2; ++j) {
                size_t uoff = ((size_t)((h * 16 + kc) * 16 + wave * 2 + j)) * FRAGB;
                load_frag(PA[j], pnA + uoff, laneB);
                load_frag(PB[j], pnB + uoff, laneB);
            }
#pragma unroll
            for (int j = 0; j < 2; ++j) vr[j] = wmma_bf16(FUr.b,  PA[j].b, vr[j]);
#pragma unroll
            for (int j = 0; j < 2; ++j) vi[j] = wmma_bf16(FUi.b,  PA[j].b, vi[j]);
#pragma unroll
            for (int j = 0; j < 2; ++j) vr[j] = wmma_bf16(FUi.b,  PB[j].b, vr[j]);
#pragma unroll
            for (int j = 0; j < 2; ++j) vi[j] = wmma_bf16(FUrn.b, PB[j].b, vi[j]);
        }

        // ========== elementwise update (diag == K/D == 2.0 exactly) =======
#pragma unroll
        for (int j = 0; j < 2; ++j) {
            int d = (wave * 2 + j) * 16 + l16;
#pragma unroll
            for (int v = 0; v < 8; ++v) {
                int s = j * 8 + v;
                int m = v + 8 * half;
                float x = xr[s], y = yr[s];
                float cr = vr[j][v] * invD - 2.0f * x;
                float ci = vi[j][v] * invD - 2.0f * y;
                float xn = x * damp + scale * cr;
                float yn = y * damp + scale * ci;
                float inv = rsqrtf(xn * xn + yn * yn + 1e-12f);
                xn *= inv; yn *= inv;
                xr[s] = xn; yr[s] = yn;
                Xbf[m * Dd + d] = (unsigned short)f2bf(xn);
                Ybf[m * Dd + d] = (unsigned short)f2bf(yn);
            }
        }
        __syncthreads();
    }

    // ================= final similarity: logits (BT,H,K) =================
    v8f dr[4], di[4];
#pragma unroll
    for (int j = 0; j < 4; ++j) { dr[j] = zero8(); di[j] = zero8(); }

#pragma unroll 1
    for (int dc = 0; dc < 8; ++dc) {
        int base = l16 * Dd + dc * 32 + half * 8;
        Frag FX, FY, FXn;
        FX.q[0] = *(const uint4*)&Xbf[base];
        FX.q[1] = *(const uint4*)&Xbf[base + 16];
        FY.q[0] = *(const uint4*)&Ybf[base];
        FY.q[1] = *(const uint4*)&Ybf[base + 16];
#pragma unroll
        for (int v = 0; v < 8; ++v) FXn.u[v] = FX.u[v] ^ 0x80008000u;

        Frag PA[4], PB[4];
#pragma unroll
        for (int j = 0; j < 4; ++j) {
            size_t uoff = ((size_t)((h * 8 + dc) * 32 + wave * 4 + j)) * FRAGB;
            load_frag(PA[j], ptA + uoff, laneB);
            load_frag(PB[j], ptB + uoff, laneB);
        }
#pragma unroll
        for (int j = 0; j < 4; ++j) dr[j] = wmma_bf16(FX.b,  PA[j].b, dr[j]);
#pragma unroll
        for (int j = 0; j < 4; ++j) di[j] = wmma_bf16(FY.b,  PA[j].b, di[j]);
#pragma unroll
        for (int j = 0; j < 4; ++j) dr[j] = wmma_bf16(FY.b,  PB[j].b, dr[j]);
#pragma unroll
        for (int j = 0; j < 4; ++j) di[j] = wmma_bf16(FXn.b, PB[j].b, di[j]);
    }
#pragma unroll
    for (int j = 0; j < 4; ++j) {
        int kt = wave * 4 + j;
#pragma unroll
        for (int v = 0; v < 8; ++v) {
            int m = v + 8 * half;
            float s = sqrtf(dr[j][v] * dr[j][v] + di[j][v] * di[j][v] + 1e-12f) * invD;
            s = fminf(fmaxf(s, 0.0f), 1.0f);
            out[(((size_t)(tokBase + m)) * Hh + h) * Kk + kt * 16 + l16] = s;
        }
    }
}

// ---------------------------------------------------------------------------
// Argmax over K per (token, head): one wave per row.
// ---------------------------------------------------------------------------
__global__ __launch_bounds__(256) void rr_argmax(const float* __restrict__ logits,
                                                 int* __restrict__ idxOut) {
    int row  = blockIdx.x * 8 + (threadIdx.x >> 5);
    int lane = threadIdx.x & 31;
    const float* p = logits + (size_t)row * Kk;
    float best = -1e30f;
    int   bi   = 0;
#pragma unroll 4
    for (int i = 0; i < Kk / 32; ++i) {
        int k = i * 32 + lane;
        float v = p[k];
        if (v > best) { best = v; bi = k; }
    }
#pragma unroll
    for (int off = 16; off >= 1; off >>= 1) {
        float ov = __shfl_xor(best, off, 32);
        int   oi = __shfl_xor(bi,   off, 32);
        if (ov > best || (ov == best && oi < bi)) { best = ov; bi = oi; }
    }
    if (lane == 0) idxOut[row] = bi;
}

// ---------------------------------------------------------------------------
extern "C" void kernel_launch(void* const* d_in, const int* in_sizes, int n_in,
                              void* d_out, int out_size, void* d_ws, size_t ws_size,
                              hipStream_t stream) {
    const float* tag = (const float*)d_in[0];       // (4,2048,256) fp32
    const float* pat = (const float*)d_in[1];       // (8,512,256) fp32
    (void)in_sizes; (void)n_in; (void)out_size; (void)ws_size;

    uint8_t* ws  = (uint8_t*)d_ws;
    uint8_t* ptA = ws + 0 * PARTB;
    uint8_t* ptB = ws + 1 * PARTB;
    uint8_t* pnA = ws + 2 * PARTB;
    uint8_t* pnB = ws + 3 * PARTB;

    float* logits = (float*)d_out;                         // BT*H*K floats
    int*   idx    = (int*)(logits + (size_t)BT * Hh * Kk); // BT*H int32

    // 1) pattern -> cos/sin bf16 fragment buffers (4 x 2048 frags x 32 lanes)
    rr_prep<<<dim3(1024), dim3(256), 0, stream>>>(pat, ws);

    // 2) main iteration: (BT/16) token tiles x H heads
    rr_main<<<dim3(BT / TM, Hh), dim3(256), 0, stream>>>(tag, ptA, ptB, pnA, pnB, logits);

    // 3) argmax over K per (token, head)
    rr_argmax<<<dim3((BT * Hh) / 8), dim3(256), 0, stream>>>(logits, idx);
}
